// SemanticEncoder_20693152432512
// MI455X (gfx1250) — matrix operand order repaired and checked by reading, not compiled, not run on hardware
//
#include <hip/hip_runtime.h>
#include <hip/hip_bf16.h>
#include <math.h>

typedef __attribute__((ext_vector_type(16))) _Float16 v16h;
typedef __attribute__((ext_vector_type(8)))  _Float16 half8;
typedef __attribute__((ext_vector_type(8)))  float    v8f;

// ---------------------------------------------------------------- helpers ---

// Map a window-ordered token index back to its (b,h,w) flat index in the
// natural BHWC layout, undoing the cyclic shift (roll by -ss before
// partition => rolled[h] = x[(h+ss) mod H]).
__device__ __forceinline__ long win_token_to_idx(int t, int H, int W, int ss) {
    int inWin = t & 63;
    int win   = t >> 6;
    int nW = W >> 3, nH = H >> 3;
    int ww = win % nW;
    int tm = win / nW;
    int wh = tm % nH;
    int b  = tm / nH;
    int r = inWin >> 3, s = inWin & 7;
    int h = wh * 8 + r, w = ww * 8 + s;
    int h0 = h + ss; if (h0 >= H) h0 -= H;
    int w0 = w + ss; if (w0 >= W) w0 -= W;
    return ((long)(b * H + h0) * W + w0);
}

// A fragment: 16x32 f16, rows striped on lanes (ISA 7.12.2 16-bit A layout).
// lane<16: row=lane, K = {0..7, 16..23}; lane>=16: row=lane-16, K={8..15,24..31}
__device__ __forceinline__ v16h load_a_frag(const _Float16* __restrict__ A,
                                            int lda, int mbase, int kbase, int lane) {
    int row = mbase + (lane & 15);
    int hi  = lane >> 4;
    const _Float16* ap = A + (long)row * lda + kbase + hi * 8;
    half8 lo = *(const half8*)(ap);
    half8 hh = *(const half8*)(ap + 16);
    v16h r;
#pragma unroll
    for (int i = 0; i < 8; ++i) { r[i] = lo[i]; r[i + 8] = hh[i]; }
    return r;
}

// B fragment: 32x16 f16, B[k][n] = W[n][k] with W row-major [N,K].
// lane<16: col=lane, K = {0..7, 16..23}; lane>=16: col=lane-16, K={8..15,24..31}
__device__ __forceinline__ v16h load_b_frag(const _Float16* __restrict__ Wm,
                                            int ldw, int nbase, int kbase, int lane) {
    int col = nbase + (lane & 15);
    int hi  = lane >> 4;
    const _Float16* wp = Wm + (long)col * ldw + kbase + hi * 8;
    half8 lo = *(const half8*)(wp);
    half8 hh = *(const half8*)(wp + 16);
    v16h r;
#pragma unroll
    for (int i = 0; i < 8; ++i) { r[i] = lo[i]; r[i + 8] = hh[i]; }
    return r;
}

__device__ __forceinline__ float gelu_exact(float v) {
    return 0.5f * v * (1.0f + erff(v * 0.70710678118654752f));
}

// Async global->LDS copy of one 16-byte chunk (CDNA5 async path, ASYNCcnt).
__device__ __forceinline__ void async_b128(unsigned lds_addr, const void* gptr) {
    asm volatile("global_load_async_to_lds_b128 %0, %1, off"
                 :: "v"(lds_addr), "v"(gptr) : "memory");
}
__device__ __forceinline__ void wait_async0() {
    asm volatile("s_wait_asynccnt 0x0" ::: "memory");
}

// ---------------------------------------------------------------- kernels ---

__global__ __launch_bounds__(256) void k_cvt_f16(const float* __restrict__ in,
                                                 _Float16* __restrict__ out, int n) {
    int i = blockIdx.x * 256 + threadIdx.x;
    if (i < n) out[i] = (_Float16)in[i];
}

// Expand the relative-position-bias table to a dense [heads][64][64] table.
__global__ __launch_bounds__(256) void k_rpb_tab(const float* __restrict__ rpb,
                                                 float* __restrict__ tab, int heads) {
    int i = blockIdx.x * 256 + threadIdx.x;
    if (i >= heads * 4096) return;
    int head = i >> 12;
    int rc = i & 4095;
    int r = rc >> 6, c = rc & 63;
    int idx = ((r >> 3) - (c >> 3) + 7) * 15 + ((r & 7) - (c & 7) + 7);
    tab[i] = rpb[idx * heads + head];
}

__global__ __launch_bounds__(256) void k_bchw_to_bhwc(const float* __restrict__ in,
                                                      float* __restrict__ out,
                                                      int B, int C, int H, int W) {
    long i = (long)blockIdx.x * 256 + threadIdx.x;
    long tot = (long)B * C * H * W;
    if (i >= tot) return;
    int c = i % C; long t = i / C;      // t = b*H*W + h*W + w
    int w = t % W; long t2 = t / W;
    int h = t2 % H; int b = t2 / H;
    out[i] = in[((long)(b * C + c) * H + h) * W + w];
}

__global__ __launch_bounds__(256) void k_bhwc_to_bchw(const float* __restrict__ in,
                                                      float* __restrict__ out,
                                                      int B, int C, int H, int W) {
    long i = (long)blockIdx.x * 256 + threadIdx.x;   // index in BCHW output
    long tot = (long)B * C * H * W;
    if (i >= tot) return;
    int w = i % W; long t = i / W;
    int h = t % H; long t2 = t / H;
    int c = t2 % C; int b = t2 / C;
    out[i] = in[((long)(b * H + h) * W + w) * C + c];
}

// Wave-per-token LayerNorm -> f16. wmap=1 gathers through the shifted-window
// permutation so GEMM rows are window-ordered tokens.
__global__ __launch_bounds__(256) void k_ln(const float* __restrict__ x,
                                            const float* __restrict__ g,
                                            const float* __restrict__ bt,
                                            _Float16* __restrict__ out,
                                            int C, int T, int H, int W, int ss, int wmap) {
    int lane = threadIdx.x & 31;
    int t = blockIdx.x * 8 + (threadIdx.x >> 5);
    if (t >= T) return;
    long base = wmap ? win_token_to_idx(t, H, W, ss) * (long)C : (long)t * C;
    float vals[12];
    float sum = 0.f, sq = 0.f;
    int nch = C >> 5;                       // C in {96,192} -> 3 or 6 chunks
#pragma unroll 4
    for (int i = 0; i < nch; ++i) {
        float v = x[base + lane + 32 * i];
        vals[i] = v; sum += v; sq += v * v;
    }
#pragma unroll
    for (int off = 16; off > 0; off >>= 1) {
        sum += __shfl_xor(sum, off, 32);
        sq  += __shfl_xor(sq,  off, 32);
    }
    float mean = sum / C;
    float inv  = rsqrtf(sq / C - mean * mean + 1e-5f);
#pragma unroll 4
    for (int i = 0; i < nch; ++i) {
        int c = lane + 32 * i;
        out[(long)t * C + c] = (_Float16)((vals[i] - mean) * inv * g[c] + bt[c]);
    }
}

// Patch-merge gather (2x2 neighborhood concat, torch order) + LN over 4*C1.
__global__ __launch_bounds__(256) void k_merge_ln(const float* __restrict__ x,
                                                  const float* __restrict__ g,
                                                  const float* __restrict__ bt,
                                                  _Float16* __restrict__ out,
                                                  int C1, int H, int W) {
    const int C4 = 4 * C1;                  // 384
    int lane = threadIdx.x & 31;
    int H2 = H >> 1, W2 = W >> 1;
    int t = blockIdx.x * 8 + (threadIdx.x >> 5);
    int w2 = t % W2; int tm = t / W2;
    int h2 = tm % H2; int b = tm / H2;
    float vals[12];
    float sum = 0.f, sq = 0.f;
    int nch = C4 >> 5;                      // 12
#pragma unroll
    for (int i = 0; i < nch; ++i) {
        int c = lane + 32 * i;
        int q = c / C1, cc = c % C1;
        int dh = q & 1, dw = q >> 1;        // order: (0,0),(1,0),(0,1),(1,1)
        float v = x[((long)(b * H + 2 * h2 + dh) * W + (2 * w2 + dw)) * C1 + cc];
        vals[i] = v; sum += v; sq += v * v;
    }
#pragma unroll
    for (int off = 16; off > 0; off >>= 1) {
        sum += __shfl_xor(sum, off, 32);
        sq  += __shfl_xor(sq,  off, 32);
    }
    float mean = sum / C4;
    float inv  = rsqrtf(sq / C4 - mean * mean + 1e-5f);
#pragma unroll
    for (int i = 0; i < nch; ++i) {
        int c = lane + 32 * i;
        out[(long)t * C4 + c] = (_Float16)((vals[i] - mean) * inv * g[c] + bt[c]);
    }
}

// Generic WMMA GEMM: Out[m,n] = sum_k A[m,k] * W[n,k] + bias[n].
// Block = 8 waves, all sharing one 32-column weight tile, async-staged into
// LDS once (ASYNCcnt path). Each wave computes a 32x32 tile: 2 A-frags +
// 2 LDS B-frags -> 4 WMMAs per K-step. Grid = (N/32, M/256).
// EPI: 0 = store f16, 1 = GELU->f16, 2 = f32 += with window-reverse scatter,
//      3 = f32 += identity, 4 = f32 write identity.
template <int EPI>
__global__ __launch_bounds__(256) void k_gemm(const _Float16* __restrict__ A, int lda,
                                              const _Float16* __restrict__ Wm,
                                              const float* __restrict__ bias,
                                              void* __restrict__ Out,
                                              int M, int N, int K,
                                              int H, int Wd, int ss) {
    __shared__ _Float16 Bsh[32 * 384];      // 24 KB max (K <= 384)

    const int tid = threadIdx.x;
    const int lane = tid & 31;
    const int wv = tid >> 5;
    const int nt = blockIdx.x;              // 32-column tile index

    // Stage weight rows [nt*32, nt*32+32) x K (contiguous in memory) to LDS.
    {
        const char* gsrc = (const char*)(Wm + (size_t)nt * 32 * K);
        unsigned lds0 = (unsigned)(uintptr_t)(void*)Bsh;
        int chunks = (32 * K * 2) >> 4;     // 16B chunks
        for (int i = tid; i < chunks; i += 256)
            async_b128(lds0 + i * 16, gsrc + i * 16);
        wait_async0();
    }
    __syncthreads();

    const int mt0 = (blockIdx.y * 8 + wv) * 2;   // first 16-row tile of wave
    v8f acc[2][2] = {};
    for (int k0 = 0; k0 < K; k0 += 32) {
        if (k0 + 32 < K)                     // prefetch next A K-chunk
            __builtin_prefetch(A + (long)(mt0 * 16 + (lane & 15)) * lda + k0 + 32, 0, 0);
        v16h a0 = load_a_frag(A, lda, mt0 * 16,      k0, lane);
        v16h a1 = load_a_frag(A, lda, mt0 * 16 + 16, k0, lane);
        v16h b0 = load_b_frag(Bsh, K, 0,  k0, lane);
        v16h b1 = load_b_frag(Bsh, K, 16, k0, lane);
        acc[0][0] = __builtin_amdgcn_wmma_f32_16x16x32_f16(false, a0, false, b0, (short)0, acc[0][0], false, false);
        acc[0][1] = __builtin_amdgcn_wmma_f32_16x16x32_f16(false, a0, false, b1, (short)0, acc[0][1], false, false);
        acc[1][0] = __builtin_amdgcn_wmma_f32_16x16x32_f16(false, a1, false, b0, (short)0, acc[1][0], false, false);
        acc[1][1] = __builtin_amdgcn_wmma_f32_16x16x32_f16(false, a1, false, b1, (short)0, acc[1][1], false, false);
    }

    const int hi = lane >> 4, ln = lane & 15;
#pragma unroll
    for (int mi = 0; mi < 2; ++mi) {
#pragma unroll
        for (int r = 0; r < 8; ++r) {
            int row = (mt0 + mi) * 16 + r + hi * 8;
#pragma unroll
            for (int ni = 0; ni < 2; ++ni) {
                int c = nt * 32 + ni * 16 + ln;
                float v = acc[mi][ni][r] + bias[c];
                if constexpr (EPI == 0) {
                    ((_Float16*)Out)[(long)row * N + c] = (_Float16)v;
                } else if constexpr (EPI == 1) {
                    ((_Float16*)Out)[(long)row * N + c] = (_Float16)gelu_exact(v);
                } else if constexpr (EPI == 2) {
                    long base = win_token_to_idx(row, H, Wd, ss) * (long)N;
                    ((float*)Out)[base + c] += v;
                } else if constexpr (EPI == 3) {
                    ((float*)Out)[(long)row * N + c] += v;
                } else {
                    ((float*)Out)[(long)row * N + c] = v;
                }
            }
        }
    }
}

// region label for the shifted-window mask (slice boundaries at L-ws, L-ss)
__device__ __forceinline__ int reg_of(int h, int L, int ss) {
    return (h < L - 8) ? 0 : (h < L - ss ? 1 : 2);
}

// Window attention for one (window, head): N=64 tokens, hd=32.
// 4 waves: S = q k^T (16 WMMA) -> LDS, pair-parallel softmax(+bias table,
// +analytic shift mask), then P @ v (16 WMMA) with v^T staged in LDS.
__global__ __launch_bounds__(128) void k_attn(const _Float16* __restrict__ qkv,
                                              const float* __restrict__ btab,
                                              _Float16* __restrict__ aout,
                                              int C, int heads, int H, int W,
                                              int ss, float scale) {
    __shared__ float    S[64 * 64];
    __shared__ _Float16 P[64 * 64];
    __shared__ _Float16 Vt[32 * 64];

    const int ld = 3 * C;
    int win  = blockIdx.x / heads;
    int head = blockIdx.x % heads;
    long winBase = (long)win * 64;
    int tid = threadIdx.x, lane = tid & 31, wv = tid >> 5;

    // window position inside image (for the shift mask)
    int nW = W >> 3, nH = H >> 3;
    int wInImg = win % (nH * nW);
    int wh = wInImg / nW, ww = wInImg % nW;

    // stage v^T into LDS: Vt[d][j] = v[token j, channel d]
    for (int i = tid; i < 64 * 32; i += 128) {
        int j = i >> 5, d = i & 31;
        Vt[d * 64 + j] = qkv[(winBase + j) * ld + 2 * C + head * 32 + d];
    }
    __syncthreads();

    // S = q k^T : each wave does a 16-row band, 4 column tiles, K=32 (1 WMMA)
    const _Float16* Aq = qkv + winBase * ld + head * 32;
    const _Float16* Kp = qkv + winBase * ld + C + head * 32;
    {
        v16h a = load_a_frag(Aq, ld, wv * 16, 0, lane);
#pragma unroll
        for (int ntb = 0; ntb < 4; ++ntb) {
            v8f acc = {};
            v16h b = load_b_frag(Kp, ld, ntb * 16, 0, lane);
            acc = __builtin_amdgcn_wmma_f32_16x16x32_f16(false, a, false, b, (short)0, acc, false, false);
            int hi = lane >> 4, ln = lane & 15;
#pragma unroll
            for (int r = 0; r < 8; ++r)
                S[(wv * 16 + r + hi * 8) * 64 + ntb * 16 + ln] = acc[r];
        }
    }
    __syncthreads();

    // softmax: 2 threads per row, register-resident, pair-combine via shfl
    {
        int i = tid >> 1;                   // row
        int j0 = (tid & 1) * 32;            // column half
        int r1 = i >> 3, s1 = i & 7;
        int lab1 = 0;
        if (ss) lab1 = reg_of(wh * 8 + r1, H, ss) * 3 + reg_of(ww * 8 + s1, W, ss);
        const float* brow = btab + head * 4096 + i * 64 + j0;
        float vloc[32];
        float mx = -3.0e38f;
#pragma unroll
        for (int jj = 0; jj < 32; ++jj) {
            int j = j0 + jj;
            float v = S[i * 64 + j] * scale + brow[jj];
            if (ss) {
                int lab2 = reg_of(wh * 8 + (j >> 3), H, ss) * 3 + reg_of(ww * 8 + (j & 7), W, ss);
                if (lab2 != lab1) v -= 100.0f;
            }
            vloc[jj] = v;
            mx = fmaxf(mx, v);
        }
        mx = fmaxf(mx, __shfl_xor(mx, 1, 32));
        float sum = 0.f;
#pragma unroll
        for (int jj = 0; jj < 32; ++jj) {
            float e = __expf(vloc[jj] - mx);
            vloc[jj] = e;
            sum += e;
        }
        sum += __shfl_xor(sum, 1, 32);
        float inv = 1.0f / sum;
#pragma unroll
        for (int jj = 0; jj < 32; ++jj)
            P[i * 64 + j0 + jj] = (_Float16)(vloc[jj] * inv);
    }
    __syncthreads();

    // out = P @ v : each wave 16 rows x 32 cols, K=64 (2 WMMA per tile)
#pragma unroll
    for (int dt = 0; dt < 2; ++dt) {
        v8f acc = {};
#pragma unroll
        for (int k0 = 0; k0 < 64; k0 += 32) {
            v16h a = load_a_frag(P, 64, wv * 16, k0, lane);
            v16h b = load_b_frag(Vt, 64, dt * 16, k0, lane);
            acc = __builtin_amdgcn_wmma_f32_16x16x32_f16(false, a, false, b, (short)0, acc, false, false);
        }
        int hi = lane >> 4, ln = lane & 15;
#pragma unroll
        for (int r = 0; r < 8; ++r) {
            int row = wv * 16 + r + hi * 8;
            aout[(winBase + row) * C + head * 32 + dt * 16 + ln] = (_Float16)acc[r];
        }
    }
}

// ------------------------------------------------------------------- host ---

struct BlkP {
    const float *mlp_b1, *mlp_b2, *mlp_w1, *mlp_w2, *n1_b, *n1_g, *n2_b, *n2_g,
                *proj_b, *proj_w, *qkv_b, *qkv_w, *rpb;
};

static BlkP blk_at(void* const* d_in, int base) {
    BlkP p;
    p.mlp_b1 = (const float*)d_in[base + 0];
    p.mlp_b2 = (const float*)d_in[base + 1];
    p.mlp_w1 = (const float*)d_in[base + 2];
    p.mlp_w2 = (const float*)d_in[base + 3];
    p.n1_b   = (const float*)d_in[base + 4];
    p.n1_g   = (const float*)d_in[base + 5];
    p.n2_b   = (const float*)d_in[base + 6];
    p.n2_g   = (const float*)d_in[base + 7];
    p.proj_b = (const float*)d_in[base + 8];
    p.proj_w = (const float*)d_in[base + 9];
    p.qkv_b  = (const float*)d_in[base + 10];
    p.qkv_w  = (const float*)d_in[base + 11];
    p.rpb    = (const float*)d_in[base + 12];
    return p;
}

static void gemm(hipStream_t s, const _Float16* A, int lda, const _Float16* Wm,
                 const float* bias, void* out, int M, int N, int K, int epi,
                 int H = 0, int Wd = 0, int ss = 0) {
    dim3 grid(N / 32, M / 256);
    switch (epi) {
    case 0: k_gemm<0><<<grid, 256, 0, s>>>(A, lda, Wm, bias, out, M, N, K, H, Wd, ss); break;
    case 1: k_gemm<1><<<grid, 256, 0, s>>>(A, lda, Wm, bias, out, M, N, K, H, Wd, ss); break;
    case 2: k_gemm<2><<<grid, 256, 0, s>>>(A, lda, Wm, bias, out, M, N, K, H, Wd, ss); break;
    case 3: k_gemm<3><<<grid, 256, 0, s>>>(A, lda, Wm, bias, out, M, N, K, H, Wd, ss); break;
    default: k_gemm<4><<<grid, 256, 0, s>>>(A, lda, Wm, bias, out, M, N, K, H, Wd, ss); break;
    }
}

extern "C" void kernel_launch(void* const* d_in, const int* in_sizes, int n_in,
                              void* d_out, int out_size, void* d_ws, size_t ws_size,
                              hipStream_t stream) {
    (void)in_sizes; (void)n_in; (void)out_size; (void)ws_size;

    // jax pytree flatten order (dict keys sorted): merge, stage1[0..1], stage2[0..1], x
    const float* m_n_b  = (const float*)d_in[0];
    const float* m_n_g  = (const float*)d_in[1];
    const float* m_redb = (const float*)d_in[2];
    const float* m_redw = (const float*)d_in[3];
    BlkP s1[2] = { blk_at(d_in, 4),  blk_at(d_in, 17) };
    BlkP s2[2] = { blk_at(d_in, 30), blk_at(d_in, 43) };
    const float* x_in = (const float*)d_in[56];

    const int B = 8;
    const int H1 = 128, W1 = 128, C1 = 96,  T1 = B * H1 * W1;
    const int H2 = 64,  W2 = 64,  C2 = 192, T2 = B * H2 * W2;
    const float scale = 0.17677669529663687f;   // 32^-0.5

    // workspace arena
    char* wp = (char*)d_ws;
    auto alloc = [&](size_t bytes) {
        char* r = wp; wp += (bytes + 255) & ~(size_t)255; return r;
    };
    float*    xA   = (float*)   alloc((size_t)T1 * C1 * 4);          // stage1 activation BHWC
    float*    xB   = (float*)   alloc((size_t)T2 * C2 * 4);          // stage2 activation BHWC
    _Float16* xln  = (_Float16*)alloc((size_t)T1 * C1 * 2);          // == T2*384*2
    _Float16* qkvb = (_Float16*)alloc((size_t)T1 * 3 * C1 * 2);
    _Float16* aob  = (_Float16*)alloc((size_t)T1 * C1 * 2);
    _Float16* hbuf = (_Float16*)alloc((size_t)T1 * 4 * C1 * 2);
    float*    btab = (float*)   alloc((size_t)6 * 4096 * 4);         // bias table (max heads=6)

    auto cvt = [&](const float* src, int n) -> _Float16* {
        _Float16* dst = (_Float16*)alloc((size_t)n * 2);
        k_cvt_f16<<<(n + 255) / 256, 256, 0, stream>>>(src, dst, n);
        return dst;
    };

    struct BlkW { _Float16 *qkv, *proj, *w1, *w2; };
    BlkW w1b[2], w2b[2];
    for (int i = 0; i < 2; ++i) {
        w1b[i].qkv  = cvt(s1[i].qkv_w,  3 * C1 * C1);
        w1b[i].proj = cvt(s1[i].proj_w, C1 * C1);
        w1b[i].w1   = cvt(s1[i].mlp_w1, 4 * C1 * C1);
        w1b[i].w2   = cvt(s1[i].mlp_w2, C1 * 4 * C1);
    }
    _Float16* redw16 = cvt(m_redw, C2 * 4 * C1);
    for (int i = 0; i < 2; ++i) {
        w2b[i].qkv  = cvt(s2[i].qkv_w,  3 * C2 * C2);
        w2b[i].proj = cvt(s2[i].proj_w, C2 * C2);
        w2b[i].w1   = cvt(s2[i].mlp_w1, 4 * C2 * C2);
        w2b[i].w2   = cvt(s2[i].mlp_w2, C2 * 4 * C2);
    }

    // input BCHW -> BHWC
    {
        long tot = (long)T1 * C1;
        k_bchw_to_bhwc<<<(int)((tot + 255) / 256), 256, 0, stream>>>(x_in, xA, B, C1, H1, W1);
    }

    auto run_block = [&](float* x, const BlkP& p, const BlkW& w,
                         int C, int heads, int H, int W, int T, int shift) {
        int ss = shift ? 4 : 0;
        // dense relative-position-bias table for this block
        k_rpb_tab<<<(heads * 4096 + 255) / 256, 256, 0, stream>>>(p.rpb, btab, heads);
        // LN1 (window-ordered gather) -> f16
        k_ln<<<T / 8, 256, 0, stream>>>(x, p.n1_g, p.n1_b, xln, C, T, H, W, ss, 1);
        // QKV
        gemm(stream, xln, C, w.qkv, p.qkv_b, qkvb, T, 3 * C, C, 0);
        // attention
        int nwin = T / 64;
        k_attn<<<nwin * heads, 128, 0, stream>>>(qkvb, btab, aob, C, heads, H, W, ss, scale);
        // proj + window-reverse + residual add into x
        gemm(stream, aob, C, w.proj, p.proj_b, x, T, C, C, 2, H, W, ss);
        // LN2 (natural order)
        k_ln<<<T / 8, 256, 0, stream>>>(x, p.n2_g, p.n2_b, xln, C, T, H, W, 0, 0);
        // MLP
        gemm(stream, xln, C, w.w1, p.mlp_b1, hbuf, T, 4 * C, C, 1);
        gemm(stream, hbuf, 4 * C, w.w2, p.mlp_b2, x, T, C, 4 * C, 3);
    };

    run_block(xA, s1[0], w1b[0], C1, 3, H1, W1, T1, 0);
    run_block(xA, s1[1], w1b[1], C1, 3, H1, W1, T1, 1);

    // patch merge: gather+LN over 4*C1, then reduce 384 -> 192 (write xB)
    k_merge_ln<<<T2 / 8, 256, 0, stream>>>(xA, m_n_g, m_n_b, xln, C1, H1, W1);
    gemm(stream, xln, 4 * C1, redw16, m_redb, xB, T2, C2, 4 * C1, 4);

    run_block(xB, s2[0], w2b[0], C2, 6, H2, W2, T2, 0);
    run_block(xB, s2[1], w2b[1], C2, 6, H2, W2, T2, 1);

    // BHWC -> BCHW into d_out
    {
        long tot = (long)T2 * C2;
        k_bhwc_to_bchw<<<(int)((tot + 255) / 256), 256, 0, stream>>>(xB, (float*)d_out, B, C2, H2, W2);
    }
}
